// Attn_Head_7722351198607
// MI455X (gfx1250) — matrix operand order repaired and checked by reading, not compiled, not run on hardware
//
#include <hip/hip_runtime.h>
#include <hip/hip_bf16.h>

typedef _Float16 v16h __attribute__((ext_vector_type(16)));
typedef _Float16 v8h  __attribute__((ext_vector_type(8)));
typedef float    v8f  __attribute__((ext_vector_type(8)));
typedef float    v4f  __attribute__((ext_vector_type(4)));

#define N_NODES 8192
#define F_IN    128
#define F_OUT   64
#define SLOPE   0.2f

// ---------------------------------------------------------------------------
// Kernel 1: one wave per node row.
//   fts[row, 0:64] = seq[row, 0:128] @ W1   (f32 accumulate)
//   f1[row] = fts . a1 + b1 ;  f2[row] = fts . a2 + b2   (wave shfl reduce)
//   ftsT[64][8192] written as f16 (transposed, so WMMA B fragments are
//   contiguous along K).
// ---------------------------------------------------------------------------
__global__ __launch_bounds__(256) void proj_kernel(
    const float* __restrict__ seq, const float* __restrict__ W1,
    const float* __restrict__ a1, const float* __restrict__ b1,
    const float* __restrict__ a2, const float* __restrict__ b2,
    _Float16* __restrict__ ftsT, float* __restrict__ f1, float* __restrict__ f2)
{
    const int wave = threadIdx.x >> 5;
    const int lane = threadIdx.x & 31;
    const int row  = blockIdx.x * 8 + wave;

    const float* srow = seq + (size_t)row * F_IN;
    const int n0 = lane, n1 = lane + 32;
    float acc0 = 0.f, acc1 = 0.f;
#pragma unroll 8
    for (int f = 0; f < F_IN; ++f) {
        float s = srow[f];
        acc0 = fmaf(s, W1[f * F_OUT + n0], acc0);
        acc1 = fmaf(s, W1[f * F_OUT + n1], acc1);
    }

    float p1 = acc0 * a1[n0] + acc1 * a1[n1];
    float p2 = acc0 * a2[n0] + acc1 * a2[n1];
#pragma unroll
    for (int o = 16; o > 0; o >>= 1) {
        p1 += __shfl_xor(p1, o, 32);
        p2 += __shfl_xor(p2, o, 32);
    }
    if (lane == 0) {
        f1[row] = p1 + b1[0];
        f2[row] = p2 + b2[0];
    }
    ftsT[(size_t)n0 * N_NODES + row] = (_Float16)acc0;
    ftsT[(size_t)n1 * N_NODES + row] = (_Float16)acc1;
}

// ---------------------------------------------------------------------------
// Kernel 2: max over f2 (monotone lrelu => row softmax max = lrelu(f1_i+maxf2))
// ---------------------------------------------------------------------------
__global__ __launch_bounds__(256) void max_kernel(
    const float* __restrict__ f2, float* __restrict__ outmax)
{
    float m = -3.4e38f;
    for (int t = threadIdx.x; t < N_NODES; t += 256) m = fmaxf(m, f2[t]);
#pragma unroll
    for (int o = 16; o > 0; o >>= 1) m = fmaxf(m, __shfl_xor(m, o, 32));
    __shared__ float red[8];
    if ((threadIdx.x & 31) == 0) red[threadIdx.x >> 5] = m;
    __syncthreads();
    if (threadIdx.x == 0) {
        float r = red[0];
#pragma unroll
        for (int i = 1; i < 8; ++i) r = fmaxf(r, red[i]);
        *outmax = r;
    }
}

// ---------------------------------------------------------------------------
// Kernel 3: fused attention. 2 waves / block; each wave owns 16 rows.
// Stream j in K=32 blocks: build probability A-tile in registers, contract
// against ftsT via v_wmma_f32_16x16x32_f16 (4 accumulators cover OUT=64).
// Epilogue: normalize, vals@Wd + bd, ELU.
// ---------------------------------------------------------------------------
__global__ __launch_bounds__(64) void attn_kernel(
    const float* __restrict__ bias, const _Float16* __restrict__ ftsT,
    const float* __restrict__ f1, const float* __restrict__ f2,
    const float* __restrict__ pmax,
    const float* __restrict__ Wd, const float* __restrict__ bdv,
    float* __restrict__ out)
{
    __shared__ float sWd[F_OUT * F_OUT];
    __shared__ float sVals[2][16 * F_OUT];
    for (int t = threadIdx.x; t < F_OUT * F_OUT; t += 64) sWd[t] = Wd[t];

    const int wave = threadIdx.x >> 5;
    const int lane = threadIdx.x & 31;
    const int M    = lane & 15;        // row within tile (A) / col within 16 (B,D)
    const int hig  = lane >> 4;        // lane half-group
    const int tile = blockIdx.x * 2 + wave;
    const int gr   = tile * 16 + M;    // global node row for this lane's A entries

    const float f1v = f1[gr];
    const float mf2 = *pmax;
    const float x0  = f1v + mf2;
    const float mi  = fmaxf(x0, SLOPE * x0);   // lrelu(f1_i + max f2) = row max

    // 16-bit A 16x32 layout: halves 0..7 -> K = kA+0..7, halves 8..15 -> K = kA+16..23
    const int kA = hig * 8;
    // 16-bit B 32x16 layout: lane = column N, halves 0..7 -> K = kB+0..7, halves 8..15 -> K = kB+16..23
    const int kB = hig * 8;

    v8f acc[4] = {v8f{}, v8f{}, v8f{}, v8f{}};
    float denom = 0.f;
    const float* brow = bias + (size_t)gr * N_NODES;

    for (int j = 0; j < N_NODES; j += 32) {
        const float* f2p = f2 + j + kA;
        v4f fa = *(const v4f*)(f2p);
        v4f fb = *(const v4f*)(f2p + 4);
        v4f fc = *(const v4f*)(f2p + 16);
        v4f fd = *(const v4f*)(f2p + 20);
        const float* bp = brow + j + kA;
        v4f qa = *(const v4f*)(bp);
        v4f qb = *(const v4f*)(bp + 4);
        v4f qc = *(const v4f*)(bp + 16);
        v4f qd = *(const v4f*)(bp + 20);

        v16h amat;
#pragma unroll
        for (int t = 0; t < 16; ++t) {
            float fv = (t < 4) ? fa[t] : (t < 8) ? fb[t - 4]
                     : (t < 12) ? fc[t - 8] : fd[t - 12];
            float bv = (t < 4) ? qa[t] : (t < 8) ? qb[t - 4]
                     : (t < 12) ? qc[t - 8] : qd[t - 12];
            float x = f1v + fv;
            float p = __expf(fmaxf(x, SLOPE * x) + bv - mi);  // exp(lrelu + bias - rowmax)
            denom  += p;
            amat[t] = (_Float16)p;
        }

        const _Float16* bbase = ftsT + j + kB;
#pragma unroll
        for (int nb = 0; nb < 4; ++nb) {
            const _Float16* p0 = bbase + (size_t)(nb * 16 + M) * N_NODES;
            v8h blo = *(const v8h*)(p0);
            v8h bhi = *(const v8h*)(p0 + 16);
            v16h bm = __builtin_shufflevector(blo, bhi,
                0, 1, 2, 3, 4, 5, 6, 7, 8, 9, 10, 11, 12, 13, 14, 15);
            acc[nb] = __builtin_amdgcn_wmma_f32_16x16x32_f16(
                false, amat, false, bm, (short)0, acc[nb], false, false);
        }
    }

    // Full row denominator: lanes L and L^16 hold disjoint K halves of row M.
    float dfull = denom + __shfl_xor(denom, 16, 32);
    const int mbase = hig * 8;   // D layout: VGPR r holds M = r (+8 for upper half)
    float* vrow = sVals[wave];
#pragma unroll
    for (int r = 0; r < 8; ++r) {
        float dr = __shfl(dfull, mbase + r, 32);   // row sum lives in lane (m)
        int m = mbase + r;
#pragma unroll
        for (int nb = 0; nb < 4; ++nb)
            vrow[m * F_OUT + nb * 16 + M] = acc[nb][r] / dr;
    }
    __syncthreads();   // sVals + sWd visible (uniform control flow across waves)

    // ret = vals @ Wd + bd ; out = elu(ret)
    const int rowbase = tile * 16;
    const float* v = sVals[wave];
    for (int idx = lane; idx < 16 * F_OUT; idx += 32) {
        int m = idx >> 6, o = idx & 63;
        float s = bdv[o];
#pragma unroll 8
        for (int n = 0; n < F_OUT; ++n)
            s = fmaf(v[m * F_OUT + n], sWd[n * F_OUT + o], s);
        out[(size_t)(rowbase + m) * F_OUT + o] = (s > 0.f) ? s : (__expf(s) - 1.f);
    }
}

// ---------------------------------------------------------------------------
extern "C" void kernel_launch(void* const* d_in, const int* in_sizes, int n_in,
                              void* d_out, int out_size, void* d_ws, size_t ws_size,
                              hipStream_t stream)
{
    const float* seq  = (const float*)d_in[0];
    const float* bias = (const float*)d_in[1];
    const float* W1   = (const float*)d_in[2];
    const float* a1   = (const float*)d_in[3];
    const float* b1   = (const float*)d_in[4];
    const float* a2   = (const float*)d_in[5];
    const float* b2   = (const float*)d_in[6];
    const float* Wd   = (const float*)d_in[7];
    const float* bdv  = (const float*)d_in[8];

    char* ws = (char*)d_ws;
    _Float16* ftsT = (_Float16*)ws;                               // 64*8192*2 = 1 MB
    float* f1   = (float*)(ws + (size_t)F_OUT * N_NODES * 2);     // 32 KB
    float* f2   = f1 + N_NODES;                                   // 32 KB
    float* pmax = f2 + N_NODES;                                   // 4 B

    proj_kernel<<<N_NODES / 8, 256, 0, stream>>>(seq, W1, a1, b1, a2, b2,
                                                 ftsT, f1, f2);
    max_kernel<<<1, 256, 0, stream>>>(f2, pmax);
    attn_kernel<<<N_NODES / 32, 64, 0, stream>>>(bias, ftsT, f1, f2, pmax,
                                                 Wd, bdv, (float*)d_out);
}
